// MS_ini_44573170598307
// MI455X (gfx1250) — matrix operand order repaired and checked by reading, not compiled, not run on hardware
//
#include <hip/hip_runtime.h>
#include <hip/hip_bf16.h>
#include <math.h>

typedef __attribute__((ext_vector_type(16))) _Float16 v16h;
typedef __attribute__((ext_vector_type(8)))  _Float16 v8h;
typedef __attribute__((ext_vector_type(8)))  float    v8f;

#define N_ROWS 8192
#define DIM    1024
#define ALPHA_ 10.0f
#define BETA_  2.0f
#define BASE_  0.5f
#define BIG_   1000000000.0f

// ---------------------------------------------------------------------------
// Pass 0: convert fp32 X (row-major 8192x1024) to f16 in workspace.
// ---------------------------------------------------------------------------
__global__ void ms_convert_f16(const float* __restrict__ x,
                               _Float16* __restrict__ xh, int n) {
  int i = blockIdx.x * blockDim.x + threadIdx.x;
  int stride = gridDim.x * blockDim.x;
  for (; i < n; i += stride) xh[i] = (_Float16)x[i];
}

// Zero the accumulated scalar outputs (loss, total) every launch.
__global__ void ms_init_out(float* __restrict__ out) {
  if (threadIdx.x == 0) {
    out[0] = 0.0f;                 // loss
    out[1 + 3 * N_ROWS] = 0.0f;    // total
  }
}

// ---------------------------------------------------------------------------
// Shared GEMM-tile helper conventions (wave32):
//   row tile rt: 16 rows, one per block.  Waves split 512 column tiles into
//   128 groups of 4; wave w handles groups w, w+8, ...
//   A frag: lane L holds row rt*16 + (L&15); halves select K sub-chunks.
//   B frag: lane L holds column ct*16 + (L&15), K chunks kb+8h..+7 and kb+16+8h..+7, as A.
//   C frag: acc[r] on lane L is sim[rt*16 + r + 8*(L>>4)][ct*16 + (L&15)].
// ---------------------------------------------------------------------------

__global__ void __launch_bounds__(256)
ms_pass1_minmax(const _Float16* __restrict__ xh,
                const int* __restrict__ targets,
                float* __restrict__ minpos, float* __restrict__ maxneg) {
  const int rt   = blockIdx.x;
  const int tid  = threadIdx.x;
  const int wave = tid >> 5;
  const int lane = tid & 31;
  const int l16  = lane & 15;
  const int h    = lane >> 4;
  const int rowBase = rt * 16;

  int tr[8];
#pragma unroll
  for (int r = 0; r < 8; ++r) tr[r] = targets[rowBase + r + 8 * h];

  float minp[8], maxn[8];
#pragma unroll
  for (int r = 0; r < 8; ++r) { minp[r] = BIG_; maxn[r] = -BIG_; }

  const _Float16* arow = xh + (size_t)(rowBase + l16) * DIM;

  for (int g = wave; g < 128; g += 8) {
    const int ct0 = g * 4;
    v8f acc[4] = {};
    int tc[4];
#pragma unroll
    for (int j = 0; j < 4; ++j) tc[j] = targets[(ct0 + j) * 16 + l16];

    for (int kb = 0; kb < DIM; kb += 32) {
      v8h c1 = *(const v8h*)(arow + kb + 8 * h);
      v8h c2 = *(const v8h*)(arow + kb + 16 + 8 * h);
      v16h a;
#pragma unroll
      for (int e = 0; e < 8; ++e) { a[e] = c1[e]; a[8 + e] = c2[e]; }
#pragma unroll
      for (int j = 0; j < 4; ++j) {
        const _Float16* bp =
            xh + (size_t)((ct0 + j) * 16 + l16) * DIM + kb + 8 * h;
        v8h d1 = *(const v8h*)bp;
        v8h d2 = *(const v8h*)(bp + 16);
        v16h b;
#pragma unroll
        for (int e = 0; e < 8; ++e) { b[e] = d1[e]; b[8 + e] = d2[e]; }
        acc[j] = __builtin_amdgcn_wmma_f32_16x16x32_f16(
            false, a, false, b, (short)0, acc[j], false, false);
      }
    }
#pragma unroll
    for (int j = 0; j < 4; ++j) {
#pragma unroll
      for (int r = 0; r < 8; ++r) {
        float s = acc[j][r];
        bool same = (tr[r] == tc[j]);
        float pv = (same && s < 1.0f) ? s : BIG_;
        float nv = (!same) ? s : -BIG_;
        minp[r] = fminf(minp[r], pv);
        maxn[r] = fmaxf(maxn[r], nv);
      }
    }
  }
  // reduce across the 16 lanes holding the same row
#pragma unroll
  for (int m = 1; m < 16; m <<= 1) {
#pragma unroll
    for (int r = 0; r < 8; ++r) {
      minp[r] = fminf(minp[r], __shfl_xor(minp[r], m, 32));
      maxn[r] = fmaxf(maxn[r], __shfl_xor(maxn[r], m, 32));
    }
  }
  __shared__ float red[8][16][2];
  if (l16 == 0) {
#pragma unroll
    for (int r = 0; r < 8; ++r) {
      red[wave][r + 8 * h][0] = minp[r];
      red[wave][r + 8 * h][1] = maxn[r];
    }
  }
  __syncthreads();
  if (tid < 16) {
    float mp = BIG_, mn = -BIG_;
#pragma unroll
    for (int w = 0; w < 8; ++w) {
      mp = fminf(mp, red[w][tid][0]);
      mn = fmaxf(mn, red[w][tid][1]);
    }
    minpos[rowBase + tid] = mp;
    maxneg[rowBase + tid] = mn;
  }
}

__global__ void __launch_bounds__(256)
ms_pass2_loss(const _Float16* __restrict__ xh,
              const int* __restrict__ targets,
              const float* __restrict__ margin,
              const float* __restrict__ minpos,
              const float* __restrict__ maxneg,
              float* __restrict__ out) {
  const int rt   = blockIdx.x;
  const int tid  = threadIdx.x;
  const int wave = tid >> 5;
  const int lane = tid & 31;
  const int l16  = lane & 15;
  const int h    = lane >> 4;
  const int rowBase = rt * 16;

  int tr[8];
  float mp[8], mn[8], mg[8];
#pragma unroll
  for (int r = 0; r < 8; ++r) {
    int row = rowBase + r + 8 * h;
    tr[r] = targets[row];
    mp[r] = minpos[row];
    mn[r] = maxneg[row];
    mg[r] = margin[row];
  }

  float apc[8] = {}, anc[8] = {}, ps[8] = {}, ns[8] = {};
  const _Float16* arow = xh + (size_t)(rowBase + l16) * DIM;

  for (int g = wave; g < 128; g += 8) {
    const int ct0 = g * 4;
    v8f acc[4] = {};
    int tc[4];
#pragma unroll
    for (int j = 0; j < 4; ++j) tc[j] = targets[(ct0 + j) * 16 + l16];

    for (int kb = 0; kb < DIM; kb += 32) {
      v8h c1 = *(const v8h*)(arow + kb + 8 * h);
      v8h c2 = *(const v8h*)(arow + kb + 16 + 8 * h);
      v16h a;
#pragma unroll
      for (int e = 0; e < 8; ++e) { a[e] = c1[e]; a[8 + e] = c2[e]; }
#pragma unroll
      for (int j = 0; j < 4; ++j) {
        const _Float16* bp =
            xh + (size_t)((ct0 + j) * 16 + l16) * DIM + kb + 8 * h;
        v8h d1 = *(const v8h*)bp;
        v8h d2 = *(const v8h*)(bp + 16);
        v16h b;
#pragma unroll
        for (int e = 0; e < 8; ++e) { b[e] = d1[e]; b[8 + e] = d2[e]; }
        acc[j] = __builtin_amdgcn_wmma_f32_16x16x32_f16(
            false, a, false, b, (short)0, acc[j], false, false);
      }
    }
#pragma unroll
    for (int j = 0; j < 4; ++j) {
#pragma unroll
      for (int r = 0; r < 8; ++r) {
        float s = acc[j][r];
        bool same = (tr[r] == tc[j]);
        bool pos = same && (s < 1.0f);
        bool neg = !same;
        bool kp = pos && (mn[r] - s + mg[r] > 0.0f);
        bool kn = neg && (s + mg[r] - mp[r] > 0.0f);
        if (kp) { apc[r] += 1.0f; ps[r] += expf(-BETA_ * (s - BASE_)); }
        if (kn) { anc[r] += 1.0f; ns[r] += expf(ALPHA_ * (s - BASE_)); }
      }
    }
  }
#pragma unroll
  for (int m = 1; m < 16; m <<= 1) {
#pragma unroll
    for (int r = 0; r < 8; ++r) {
      apc[r] += __shfl_xor(apc[r], m, 32);
      anc[r] += __shfl_xor(anc[r], m, 32);
      ps[r]  += __shfl_xor(ps[r],  m, 32);
      ns[r]  += __shfl_xor(ns[r],  m, 32);
    }
  }
  __shared__ float red[8][16][4];
  if (l16 == 0) {
#pragma unroll
    for (int r = 0; r < 8; ++r) {
      red[wave][r + 8 * h][0] = apc[r];
      red[wave][r + 8 * h][1] = anc[r];
      red[wave][r + 8 * h][2] = ps[r];
      red[wave][r + 8 * h][3] = ns[r];
    }
  }
  __syncthreads();
  if (tid < 16) {
    float ap = 0.f, an = 0.f, psum = 0.f, nsum = 0.f;
#pragma unroll
    for (int w = 0; w < 8; ++w) {
      ap += red[w][tid][0];
      an += red[w][tid][1];
      psum += red[w][tid][2];
      nsum += red[w][tid][3];
    }
    bool valid = (ap > 0.5f) && (an > 0.5f);
    float pl = (2.0f / BETA_) * log1pf(psum);
    float nl = (2.0f / ALPHA_) * log1pf(nsum);
    int row = rowBase + tid;
    out[1 + row]              = valid ? 1.0f : 0.0f;   // anchor
    out[1 + N_ROWS + row]     = valid ? ap : 0.0f;     // ap
    out[1 + 2 * N_ROWS + row] = valid ? an : 0.0f;     // an
    if (valid) {
      atomicAdd(&out[0], (pl + nl) * (1.0f / (float)N_ROWS));  // loss
      atomicAdd(&out[1 + 3 * N_ROWS], ap + an);                // total
    }
  }
}

extern "C" void kernel_launch(void* const* d_in, const int* in_sizes, int n_in,
                              void* d_out, int out_size, void* d_ws,
                              size_t ws_size, hipStream_t stream) {
  const float* x       = (const float*)d_in[0];
  const int*   targets = (const int*)d_in[1];
  const float* margin  = (const float*)d_in[2];
  float* out = (float*)d_out;

  _Float16* xh = (_Float16*)d_ws;
  float* minpos = (float*)((char*)d_ws + (size_t)N_ROWS * DIM * sizeof(_Float16));
  float* maxneg = minpos + N_ROWS;

  ms_convert_f16<<<2048, 256, 0, stream>>>(x, xh, N_ROWS * DIM);
  ms_init_out<<<1, 32, 0, stream>>>(out);
  ms_pass1_minmax<<<N_ROWS / 16, 256, 0, stream>>>(xh, targets, minpos, maxneg);
  ms_pass2_loss<<<N_ROWS / 16, 256, 0, stream>>>(xh, targets, margin, minpos,
                                                 maxneg, out);
}
